// BiLSTM_CRF_53944789238068
// MI455X (gfx1250) — matrix operand order repaired and checked by reading, not compiled, not run on hardware
//
#include <hip/hip_runtime.h>
#include <cstdint>
#include <cstddef>

typedef __attribute__((ext_vector_type(8)))  float  v8f;
typedef __attribute__((ext_vector_type(8)))  __bf16 v8bf;
typedef __attribute__((ext_vector_type(16))) __bf16 v16bf;
typedef __attribute__((ext_vector_type(4)))  int    v4i;

// AS-qualified pointer types for the gfx1250 async global->LDS builtin:
// signature (from hipcc diagnostic): (AS1 int4*, AS3 int4*, imm, imm)
typedef __attribute__((address_space(1))) v4i* gptr4;
typedef __attribute__((address_space(3))) v4i* lptr4;

// Problem constants (fixed-shape task)
constexpr int Ssz  = 512;          // sequence length
constexpr int Bsz  = 64;           // batch
constexpr int Esz  = 256;          // embedding dim
constexpr int Hsz  = 256;          // hidden per direction
constexpr int G4H  = 1024;         // 4*H (gates)
constexpr int HID2 = 512;          // 2*H (bidir concat)
constexpr int SB   = Ssz * Bsz;    // 32768 rows
#define NTAG      9
#define START_TAG 7
#define STOP_TAG  8
#define NEGV      (-100000.0f)

#if __has_builtin(__builtin_amdgcn_global_load_async_to_lds_b128) && \
    __has_builtin(__builtin_amdgcn_s_wait_asynccnt)
#define HAVE_ASYNC_LDS 1
#else
#define HAVE_ASYNC_LDS 0
#endif

// ---------------- WMMA fragment loaders (CDNA5 16x16x32 bf16 layouts) -------

// A-matrix 16x32 bf16, row-major source with leading dim lda:
// lanes 0-15: row M=lane, K k0..k0+15; lanes 16-31: row M=lane-16, K k0+16..31
__device__ __forceinline__ v16bf load_a_frag(const __bf16* A, int lda, int m0,
                                             int k0, int lane) {
  const int r  = m0 + (lane & 15);
  const int kk = k0 + ((lane >> 4) << 4);
  return *(const v16bf*)(A + (size_t)r * lda + kk);
}

// B-matrix 32x16 (B[k][n] = W[n][k], W row-major (N,K) with leading dim ldw):
// lanes 0-15: col N=lane, K k0..k0+15; lanes 16-31: col N=lane-16, K k0+16..31
__device__ __forceinline__ v16bf load_b_frag(const __bf16* W, int ldw, int n0,
                                             int k0, int lane) {
  const int n  = n0 + (lane & 15);
  const int kk = k0 + ((lane >> 4) << 4);
  return *(const v16bf*)(W + (size_t)n * ldw + kk);
}

__device__ __forceinline__ float sigmf(float x) {
  return 1.0f / (1.0f + __expf(-x));
}

// ---------------- Utility kernels -------------------------------------------

__global__ void k_cast_bf16(const float* __restrict__ in,
                            __bf16* __restrict__ out, int n) {
  int i = blockIdx.x * blockDim.x + threadIdx.x;
  if (i < n) out[i] = (__bf16)in[i];
}

__global__ void k_bias_sum(const float* __restrict__ a,
                           const float* __restrict__ b,
                           float* __restrict__ out, int n) {
  int i = blockIdx.x * blockDim.x + threadIdx.x;
  if (i < n) out[i] = a[i] + b[i];
}

// Embedding gather, f32 -> bf16, laid out (S, B, E): row = s*B + b
__global__ void k_embed(const int* __restrict__ sent,
                        const float* __restrict__ emb,
                        __bf16* __restrict__ X) {
  size_t idx = (size_t)blockIdx.x * blockDim.x + threadIdx.x;
  if (idx >= (size_t)SB * Esz) return;
  const int e   = (int)(idx & (Esz - 1));
  const int row = (int)(idx >> 8);
  const int b   = row & (Bsz - 1);
  const int s   = row >> 6;
  const int v   = sent[b * Ssz + s];
  X[idx] = (__bf16)emb[(size_t)v * Esz + e];
}

// ---------------- Batched input-projection GEMM (WMMA bf16) -----------------
// C(M,N) bf16 = A(M,K) bf16 @ W(N,K)^T bf16 + bias(N) f32, f32 accumulate.
// Block = 256 threads = 8 waves; wave computes one 16-row M-tile x 4 N-tiles.
// The 64-wide B panel is staged into LDS via async global->LDS DMA, double
// buffered: one 64x32 k-chunk (4 KB = 256 x b128, one per thread) in flight
// while WMMAs consume the previous chunk.
__global__ void __launch_bounds__(256) k_gemm(const __bf16* __restrict__ A,
                                              const __bf16* __restrict__ W,
                                              const float* __restrict__ bias,
                                              __bf16* __restrict__ C, int M,
                                              int N, int K) {
  const int tid  = threadIdx.x;
  const int wave = tid >> 5;
  const int lane = tid & 31;
  const int ln   = lane & 15;
  const int lh   = lane >> 4;
  const int m0   = (blockIdx.y * 8 + wave) * 16;
  const int n0   = blockIdx.x * 64;

  __shared__ __bf16 Bsh[2][64 * 32];  // 8 KB double buffer

  // Staging coords: thread -> (row within panel, 16B chunk within 32-col row)
  const int sn = tid >> 2;
  const int sc = (tid & 3) << 3;
  const __bf16* Wrow = W + (size_t)(n0 + sn) * K + sc;

  auto stage = [&](int buf, int k0) {
    const __bf16* src = Wrow + k0;
    __bf16* dst = &Bsh[buf][sn * 32 + sc];
#if HAVE_ASYNC_LDS
    __builtin_amdgcn_global_load_async_to_lds_b128((gptr4)src, (lptr4)dst, 0,
                                                   0);
#else
    *(v8bf*)dst = *(const v8bf*)src;
#endif
  };

  v8f acc[4];
#pragma unroll
  for (int j = 0; j < 4; ++j) {
    const float bj = bias[n0 + j * 16 + ln];
#pragma unroll
    for (int r = 0; r < 8; ++r) acc[j][r] = bj;
  }

  stage(0, 0);
  int it = 0;
  for (int k0 = 0; k0 < K; k0 += 32, ++it) {
    const int  cur  = it & 1;
    const bool more = (k0 + 32 < K);  // block-uniform
    if (more) {
      stage(cur ^ 1, k0 + 32);
#if HAVE_ASYNC_LDS
      __builtin_amdgcn_s_wait_asynccnt(1);  // drain `cur`, keep next in flight
#endif
    } else {
#if HAVE_ASYNC_LDS
      __builtin_amdgcn_s_wait_asynccnt(0);
#endif
    }
    __syncthreads();  // all waves' DMA for `cur` visible

    const v16bf a = load_a_frag(A, K, m0, k0, lane);
#pragma unroll
    for (int j = 0; j < 4; ++j) {
      const v16bf bb = *(const v16bf*)(&Bsh[cur][(j * 16 + ln) * 32 + lh * 16]);
      acc[j] = __builtin_amdgcn_wmma_f32_16x16x32_bf16(false, a, false, bb,
                                                       (short)0, acc[j], false,
                                                       false);
    }
    __syncthreads();  // readers done before `cur` is re-staged next iter
  }

#pragma unroll
  for (int j = 0; j < 4; ++j)
#pragma unroll
    for (int r = 0; r < 8; ++r) {
      const int m = m0 + r + lh * 8;
      const int n = n0 + j * 16 + ln;
      C[(size_t)m * N + n] = (__bf16)acc[j][r];
    }
}

// ---------------- Persistent recurrent LSTM kernel --------------------------
// One workgroup per direction (blockIdx.x = dir; dir 1 runs reversed).
// 1024 threads = 32 waves. h (64x256 bf16) lives in LDS; c lives in registers.
// Wave w: m_tile = w&3 (16 batch rows), hidden tiles {2*(w>>2), 2*(w>>2)+1};
// each hidden tile accumulates all 4 gates (i,f,g,o) so the nonlinearity
// fuses entirely in registers. Per step: 8 k-steps x 8 WMMA per wave.
__global__ void __launch_bounds__(1024) k_lstm(
    const __bf16* __restrict__ Gx0, const __bf16* __restrict__ Gx1,
    const __bf16* __restrict__ WhhAll,  // (2, 1024, 256) bf16
    __bf16* __restrict__ Hout) {        // (S, B, 512) bf16
  const int dir    = blockIdx.x;
  const __bf16* Gx = dir ? Gx1 : Gx0;
  const __bf16* W  = WhhAll + (size_t)dir * G4H * Hsz;
  const int colOff = dir * Hsz;

  __shared__ __bf16 h_lds[Bsz * Hsz];  // 32 KB

  const int tid  = threadIdx.x;
  const int wave = tid >> 5;
  const int lane = tid & 31;
  const int ln   = lane & 15;
  const int lh   = lane >> 4;
  const int m0   = (wave & 3) << 4;
  const int ht2  = wave >> 2;  // 0..7

  for (int i = tid; i < Bsz * Hsz; i += 1024) h_lds[i] = (__bf16)0.0f;
  float cst[2][8];
#pragma unroll
  for (int a = 0; a < 2; ++a)
#pragma unroll
    for (int r = 0; r < 8; ++r) cst[a][r] = 0.0f;
  __syncthreads();

  for (int t = 0; t < Ssz; ++t) {
    const int tt = dir ? (Ssz - 1 - t) : t;
    const __bf16* grow = Gx + (size_t)tt * Bsz * G4H;

    // Init accumulators from precomputed x@W_ih^T + bias (bf16 -> f32)
    v8f acc[2][4];
#pragma unroll
    for (int hti = 0; hti < 2; ++hti) {
      const int htAbs = ht2 * 2 + hti;
#pragma unroll
      for (int q = 0; q < 4; ++q) {
        const int n = q * Hsz + htAbs * 16 + ln;
#pragma unroll
        for (int r = 0; r < 8; ++r) {
          const int m = m0 + r + lh * 8;
          acc[hti][q][r] = (float)grow[(size_t)m * G4H + n];
        }
      }
    }

    // Prefetch next timestep's 128 KB Gx slab while WMMAs run
    // (lowers to global_prefetch_b8; 1024 threads x one cacheline each).
    if (t + 1 < Ssz) {
      const int ttn = dir ? (Ssz - 2 - t) : (t + 1);
      __builtin_prefetch(Gx + (size_t)ttn * Bsz * G4H + (size_t)tid * 64, 0, 1);
    }

    // G += h @ W_hh^T  (A-frag from LDS reused across all 8 tiles)
    for (int k0 = 0; k0 < Hsz; k0 += 32) {
      const v16bf a = load_a_frag(h_lds, Hsz, m0, k0, lane);
#pragma unroll
      for (int hti = 0; hti < 2; ++hti) {
        const int htAbs = ht2 * 2 + hti;
#pragma unroll
        for (int q = 0; q < 4; ++q) {
          const v16bf bb = load_b_frag(W, Hsz, q * Hsz + htAbs * 16, k0, lane);
          acc[hti][q] = __builtin_amdgcn_wmma_f32_16x16x32_bf16(
              false, a, false, bb, (short)0, acc[hti][q], false, false);
        }
      }
    }
    __syncthreads();  // all reads of old h done before anyone overwrites

    // Fused gates; c stays in registers; h -> LDS + global
#pragma unroll
    for (int hti = 0; hti < 2; ++hti) {
      const int htAbs = ht2 * 2 + hti;
      const int hcol  = htAbs * 16 + ln;
#pragma unroll
      for (int r = 0; r < 8; ++r) {
        const int m    = m0 + r + lh * 8;
        const float gi = sigmf(acc[hti][0][r]);
        const float gf = sigmf(acc[hti][1][r]);
        const float gg = tanhf(acc[hti][2][r]);
        const float go = sigmf(acc[hti][3][r]);
        const float cn = gf * cst[hti][r] + gi * gg;
        cst[hti][r] = cn;
        const float hv = go * tanhf(cn);
        const __bf16 hb = (__bf16)hv;
        h_lds[m * Hsz + hcol] = hb;
        Hout[((size_t)tt * Bsz + m) * HID2 + colOff + hcol] = hb;
      }
    }
    __syncthreads();  // new h visible for next step
  }
}

// ---------------- FC to tag space (tiny: T=9) -------------------------------
__global__ void k_fc(const __bf16* __restrict__ H, const float* __restrict__ Wf,
                     const float* __restrict__ bf, float* __restrict__ F) {
  int idx = blockIdx.x * blockDim.x + threadIdx.x;
  if (idx >= SB * NTAG) return;
  const int t   = idx % NTAG;
  const int row = idx / NTAG;
  const __bf16* hr = H + (size_t)row * HID2;
  const float*  wr = Wf + (size_t)t * HID2;
  float s = bf[t];
  for (int k = 0; k < HID2; ++k) s += (float)hr[k] * wr[k];
  F[idx] = s;  // (s*B+b)*9 + t
}

// ---------------- CRF: forward logsumexp + gold score, one wave per batch ---
__global__ void k_crf(const float* __restrict__ F, const int* __restrict__ tags,
                      const int* __restrict__ lens,
                      const float* __restrict__ tr,
                      float* __restrict__ score) {
  const int b    = blockIdx.x;
  const int lane = threadIdx.x;
  const int len  = lens[b];
  const bool act = lane < NTAG;
  const int  j   = act ? lane : 0;

  float trj[NTAG];
#pragma unroll
  for (int i = 0; i < NTAG; ++i) trj[i] = tr[i * NTAG + j];

  float alpha = (act && lane == START_TAG) ? 0.0f : NEGV;

  for (int t = 0; t < Ssz; ++t) {
    float av[NTAG];
#pragma unroll
    for (int i = 0; i < NTAG; ++i) av[i] = __shfl(alpha, i, 32);
    float mx = -3.0e38f;
#pragma unroll
    for (int i = 0; i < NTAG; ++i) mx = fmaxf(mx, av[i] + trj[i]);
    float ss = 0.0f;
#pragma unroll
    for (int i = 0; i < NTAG; ++i) ss += __expf(av[i] + trj[i] - mx);
    const float ft = F[((size_t)t * Bsz + b) * NTAG + j];
    const float na = mx + __logf(ss) + ft;
    if (act && t < len) alpha = na;
  }

  float val = act ? (alpha + tr[j * NTAG + STOP_TAG]) : NEGV;
  float mx2 = -3.0e38f;
#pragma unroll
  for (int i = 0; i < NTAG; ++i) mx2 = fmaxf(mx2, __shfl(val, i, 32));
  float ss2 = 0.0f;
#pragma unroll
  for (int i = 0; i < NTAG; ++i) ss2 += __expf(__shfl(val, i, 32) - mx2);
  const float fwd = mx2 + __logf(ss2);

  if (lane == 0) {
    const int tag0 = tags[b * Ssz + 0];
    float gold =
        tr[START_TAG * NTAG + tag0] + F[((size_t)0 * Bsz + b) * NTAG + tag0];
    for (int s = 0; s + 1 < Ssz; ++s) {
      if (s + 1 < len) {
        const int ta = tags[b * Ssz + s];
        const int tb = tags[b * Ssz + s + 1];
        gold += tr[ta * NTAG + tb] + F[((size_t)s * Bsz + b) * NTAG + tb];
      }
    }
    const int lastT = tags[b * Ssz + (len - 1)];
    gold += tr[lastT * NTAG + STOP_TAG];
    score[b] = fwd - gold;
  }
}

// Fixed-order reduce (deterministic; no float atomics)
__global__ void k_reduce(const float* __restrict__ score,
                         float* __restrict__ out) {
  float s = 0.0f;
  for (int b = 0; b < Bsz; ++b) s += score[b];
  out[0] = s / (float)Bsz;
}

// ---------------- Host orchestration ----------------------------------------
extern "C" void kernel_launch(void* const* d_in, const int* in_sizes, int n_in,
                              void* d_out, int out_size, void* d_ws,
                              size_t ws_size, hipStream_t stream) {
  (void)in_sizes; (void)n_in; (void)out_size; (void)ws_size;
  const int*   sent  = (const int*)d_in[0];
  const int*   tags  = (const int*)d_in[1];
  const int*   lens  = (const int*)d_in[2];
  const float* emb   = (const float*)d_in[3];
  const float* wih0  = (const float*)d_in[4];
  const float* whh0  = (const float*)d_in[5];
  const float* bih0  = (const float*)d_in[6];
  const float* bhh0  = (const float*)d_in[7];
  const float* wih1  = (const float*)d_in[8];
  const float* whh1  = (const float*)d_in[9];
  const float* bih1  = (const float*)d_in[10];
  const float* bhh1  = (const float*)d_in[11];
  const float* fcw   = (const float*)d_in[12];
  const float* fcb   = (const float*)d_in[13];
  const float* trans = (const float*)d_in[14];

  char*  base = (char*)d_ws;
  size_t off  = 0;
  auto alloc = [&](size_t bytes) -> void* {
    void* p = base + off;
    off = (off + bytes + 255) & ~(size_t)255;
    return p;
  };

  __bf16* X     = (__bf16*)alloc((size_t)SB * Esz * 2);       // 16 MiB
  __bf16* Wih0b = (__bf16*)alloc((size_t)2 * G4H * Esz * 2);  // 1 MiB
  __bf16* Whh0b = (__bf16*)alloc((size_t)2 * G4H * Hsz * 2);  // 1 MiB
  __bf16* Wih1b = (__bf16*)alloc((size_t)2 * G4H * HID2 * 2); // 2 MiB
  __bf16* Whh1b = (__bf16*)alloc((size_t)2 * G4H * Hsz * 2);  // 1 MiB
  float*  B0    = (float*)alloc((size_t)2 * G4H * 4);
  float*  B1    = (float*)alloc((size_t)2 * G4H * 4);
  __bf16* Gx0   = (__bf16*)alloc((size_t)SB * G4H * 2);       // 64 MiB (reused)
  __bf16* Gx1   = (__bf16*)alloc((size_t)SB * G4H * 2);       // 64 MiB (reused)
  __bf16* H0    = (__bf16*)alloc((size_t)SB * HID2 * 2);      // 32 MiB
  __bf16* H1    = (__bf16*)alloc((size_t)SB * HID2 * 2);      // 32 MiB
  float*  Fb    = (float*)alloc((size_t)SB * NTAG * 4);       // 1.2 MiB
  float*  Sc    = (float*)alloc((size_t)Bsz * 4);

  auto up = [](int n) { return (n + 255) / 256; };

  // Weights -> bf16, combined biases
  k_cast_bf16<<<up(2 * G4H * Esz), 256, 0, stream>>>(wih0, Wih0b, 2 * G4H * Esz);
  k_cast_bf16<<<up(2 * G4H * Hsz), 256, 0, stream>>>(whh0, Whh0b, 2 * G4H * Hsz);
  k_cast_bf16<<<up(2 * G4H * HID2), 256, 0, stream>>>(wih1, Wih1b,
                                                      2 * G4H * HID2);
  k_cast_bf16<<<up(2 * G4H * Hsz), 256, 0, stream>>>(whh1, Whh1b, 2 * G4H * Hsz);
  k_bias_sum<<<up(2 * G4H), 256, 0, stream>>>(bih0, bhh0, B0, 2 * G4H);
  k_bias_sum<<<up(2 * G4H), 256, 0, stream>>>(bih1, bhh1, B1, 2 * G4H);

  // Embedding gather (S,B,E) bf16
  {
    size_t n = (size_t)SB * Esz;
    k_embed<<<(unsigned)((n + 255) / 256), 256, 0, stream>>>(sent, emb, X);
  }

  const dim3 gemmGrid(G4H / 64, SB / 128);

  // Layer 0: input projections (both dirs), then persistent recurrence
  k_gemm<<<gemmGrid, 256, 0, stream>>>(X, Wih0b, B0, Gx0, SB, G4H, Esz);
  k_gemm<<<gemmGrid, 256, 0, stream>>>(X, Wih0b + (size_t)G4H * Esz, B0 + G4H,
                                       Gx1, SB, G4H, Esz);
  k_lstm<<<2, 1024, 0, stream>>>(Gx0, Gx1, Whh0b, H0);

  // Layer 1: input projections from H0 (K=512), recurrence (Gx buffers reused)
  k_gemm<<<gemmGrid, 256, 0, stream>>>(H0, Wih1b, B1, Gx0, SB, G4H, HID2);
  k_gemm<<<gemmGrid, 256, 0, stream>>>(H0, Wih1b + (size_t)G4H * HID2, B1 + G4H,
                                       Gx1, SB, G4H, HID2);
  k_lstm<<<2, 1024, 0, stream>>>(Gx0, Gx1, Whh1b, H1);

  // FC + CRF + deterministic reduce
  k_fc<<<up(SB * NTAG), 256, 0, stream>>>(H1, fcw, fcb, Fb);
  k_crf<<<Bsz, 32, 0, stream>>>(Fb, tags, lens, trans, Sc);
  k_reduce<<<1, 1, 0, stream>>>(Sc, (float*)d_out);
}
